// CrossGraphAttentionModel_5446018532037
// MI455X (gfx1250) — matrix operand order repaired and checked by reading, not compiled, not run on hardware
//
#include <hip/hip_runtime.h>

#define NMOL  4096
#define NPROT 4096
#define EMOL  65536
#define EPROT 65536
#define HDIM  256
#define NHEADS 4
#define HD    64
#define NG    32

typedef __attribute__((ext_vector_type(16))) _Float16 v16h;
typedef __attribute__((ext_vector_type(8)))  _Float16 v8h;
typedef __attribute__((ext_vector_type(8)))  float    v8f;

__device__ __forceinline__ v8f wmma_f16(v16h a, v16h b, v8f c) {
  // D = A(16x32 f16) * B(32x16 f16) + C(16x16 f32)
  return __builtin_amdgcn_wmma_f32_16x16x32_f16(false, a, false, b, (short)0, c, false, false);
}

__device__ __forceinline__ void atomic_add_f32(float* p, float v) {
  __hip_atomic_fetch_add(p, v, __ATOMIC_RELAXED, __HIP_MEMORY_SCOPE_AGENT);
}

// ---------------- utility kernels ----------------

__global__ void k_copy4(const float* __restrict__ src, float* __restrict__ dst, int n4) {
  int i = blockIdx.x * blockDim.x + threadIdx.x;
  if (i < n4) ((float4*)dst)[i] = ((const float4*)src)[i];
}

__global__ void k_zero(float* __restrict__ p, int n) {
  int i = blockIdx.x * blockDim.x + threadIdx.x;
  if (i < n) p[i] = 0.0f;
}

// repack f32 W[K][ncols] into f16 WMMA B-fragment order:
// Wp[((kt*ctiles+ct)*32 + lane)*16 + i] = W[kt*32 + (i<8 ? (lane>>4)*8 + i : 16 + (lane>>4)*8 + i-8)][ct*16 + (lane&15)]
__global__ void k_pack_w(const float* __restrict__ W, _Float16* __restrict__ Wp,
                         int K, int ncols) {
  int idx = blockIdx.x * blockDim.x + threadIdx.x;
  if (idx >= K * ncols) return;
  int i = idx & 15;
  int lane = (idx >> 4) & 31;
  int rest = idx >> 9;
  int ctiles = ncols >> 4;
  int ct = rest % ctiles;
  int kt = rest / ctiles;
  int kk = kt * 32 + (i < 8 ? (lane >> 4) * 8 + i : 16 + (lane >> 4) * 8 + (i - 8));
  int cc = ct * 16 + (lane & 15);
  Wp[idx] = (_Float16)W[(size_t)kk * ncols + cc];
}

// y[n, 256] = x[n, din] @ w[din, 256] + b   (din is tiny: 8 or 4)
__global__ void k_node_embed(const float* __restrict__ x, const float* __restrict__ w,
                             const float* __restrict__ b, float* __restrict__ y,
                             int n, int din) {
  int idx = blockIdx.x * blockDim.x + threadIdx.x;
  int node = idx >> 8, f = idx & 255;
  if (node >= n) return;
  float s = b[f];
  for (int k = 0; k < din; ++k) s += x[node * din + k] * w[k * HDIM + f];
  y[(size_t)node * HDIM + f] = s;
}

// msg = relu(x[src] + (ea @ ew + eb)) ; agg[dst] += msg.  4 features per thread (float4).
__global__ void k_edge_msg(const float* __restrict__ x, const float* __restrict__ ea,
                           const int* __restrict__ ei, const float* __restrict__ ew,
                           const float* __restrict__ eb, float* __restrict__ agg, int E) {
  int idx = blockIdx.x * blockDim.x + threadIdx.x;
  int e = idx >> 6, q = (idx & 63) << 2;
  if (e >= E) return;
  int src = ei[e], dst = ei[E + e];
  float a0 = ea[e * 2 + 0], a1 = ea[e * 2 + 1];
  float4 w0 = *(const float4*)(ew + q);
  float4 w1 = *(const float4*)(ew + HDIM + q);
  float4 bb = *(const float4*)(eb + q);
  float4 xv = *(const float4*)(x + (size_t)src * HDIM + q);
  float* ag = agg + (size_t)dst * HDIM + q;
  atomic_add_f32(ag + 0, fmaxf(xv.x + a0 * w0.x + a1 * w1.x + bb.x, 0.0f));
  atomic_add_f32(ag + 1, fmaxf(xv.y + a0 * w0.y + a1 * w1.y + bb.y, 0.0f));
  atomic_add_f32(ag + 2, fmaxf(xv.z + a0 * w0.z + a1 * w1.z + bb.z, 0.0f));
  atomic_add_f32(ag + 3, fmaxf(xv.w + a0 * w0.w + a1 * w1.w + bb.w, 0.0f));
}

// ---------------- WMMA GEMM ----------------
// Y = act(X[nrows,K] @ W[K,ncols] + bias); W pre-packed f16 fragments.
// One wave computes a 16x64 tile (A fragment reused across 4 column tiles).
// omode: 0 = f32 out, 1 = f16 row-major out, 2 = f16 transposed out (Yh[col*nrows+row])
__global__ void k_gemm(const float* __restrict__ X, const _Float16* __restrict__ Wp,
                       const float* __restrict__ bias, float* __restrict__ Yf,
                       _Float16* __restrict__ Yh, int nrows, int K, int ncols,
                       int do_relu, int omode) {
  const int lane = threadIdx.x & 31;
  const int wv = threadIdx.x >> 5;
  const int lo = lane & 15, hi = lane >> 4;
  const int cblocks = ncols >> 6;
  const int ctiles = ncols >> 4;
  int gw = blockIdx.x * (blockDim.x >> 5) + wv;
  int total = (nrows >> 4) * cblocks;
  if (gw >= total) return;
  int rt = gw / cblocks, cb = gw % cblocks;
  const float* Xr = X + (size_t)(rt * 16 + lo) * K;

  v8f acc[4];
#pragma unroll
  for (int c = 0; c < 4; ++c)
#pragma unroll
    for (int r = 0; r < 8; ++r) acc[c][r] = 0.0f;

#pragma unroll 2
  for (int k0 = 0; k0 < K; k0 += 32) {
    // A fragment: two contiguous 8-float runs per lane -> 4x b128 + packed cvt
    float4 xa = *(const float4*)(Xr + k0 + hi * 8);
    float4 xb = *(const float4*)(Xr + k0 + hi * 8 + 4);
    float4 xc = *(const float4*)(Xr + k0 + 16 + hi * 8);
    float4 xd = *(const float4*)(Xr + k0 + 16 + hi * 8 + 4);
    // all 4 B fragments issued before any WMMA so the loads overlap
    const _Float16* wbase = Wp + ((size_t)((k0 >> 5) * ctiles + cb * 4) * 32 + lane) * 16;
    v16h b0 = *(const v16h*)(wbase);
    v16h b1 = *(const v16h*)(wbase + 512);
    v16h b2 = *(const v16h*)(wbase + 1024);
    v16h b3 = *(const v16h*)(wbase + 1536);
    v16h a;
    a[0] = (_Float16)xa.x; a[1] = (_Float16)xa.y; a[2] = (_Float16)xa.z; a[3] = (_Float16)xa.w;
    a[4] = (_Float16)xb.x; a[5] = (_Float16)xb.y; a[6] = (_Float16)xb.z; a[7] = (_Float16)xb.w;
    a[8] = (_Float16)xc.x; a[9] = (_Float16)xc.y; a[10] = (_Float16)xc.z; a[11] = (_Float16)xc.w;
    a[12] = (_Float16)xd.x; a[13] = (_Float16)xd.y; a[14] = (_Float16)xd.z; a[15] = (_Float16)xd.w;
    acc[0] = wmma_f16(a, b0, acc[0]);
    acc[1] = wmma_f16(a, b1, acc[1]);
    acc[2] = wmma_f16(a, b2, acc[2]);
    acc[3] = wmma_f16(a, b3, acc[3]);
  }

#pragma unroll
  for (int c = 0; c < 4; ++c) {
    int ct = cb * 4 + c;
    int col = ct * 16 + lo;
    float bb = bias[col];
    if (omode == 0) {
#pragma unroll
      for (int r = 0; r < 8; ++r) {
        float v = acc[c][r] + bb;
        if (do_relu) v = fmaxf(v, 0.0f);
        Yf[(size_t)(rt * 16 + r + 8 * hi) * ncols + col] = v;
      }
    } else if (omode == 1) {
#pragma unroll
      for (int r = 0; r < 8; ++r)
        Yh[(size_t)(rt * 16 + r + 8 * hi) * ncols + col] = (_Float16)(acc[c][r] + bb);
    } else {
      v8h hv;
#pragma unroll
      for (int r = 0; r < 8; ++r) hv[r] = (_Float16)(acc[c][r] + bb);
      *(v8h*)(Yh + (size_t)col * nrows + rt * 16 + 8 * hi) = hv;  // one b128 store
    }
  }
}

// ---------------- streaming (flash) cross-attention ----------------
// Q,K: f16 row-major [n,256]; Vt: f16 transposed [256,nk]. Out = Hres + softmax(QK^T/8)V.
// One wave per (head, 16-query tile). Wave-private LDS swizzle (no block barriers):
// LDS ops from one wave retire in order (DScnt), so store->load within the wave is safe.
__global__ void k_attn(const _Float16* __restrict__ Q, const _Float16* __restrict__ Kf,
                       const _Float16* __restrict__ Vt, const float* __restrict__ Hres,
                       float* __restrict__ Out, int nq, int nk) {
  __shared__ _Float16 plds[4][16 * 32];
  const int lane = threadIdx.x & 31;
  const int wv = threadIdx.x >> 5;
  const int lo = lane & 15, hi = lane >> 4;
  int gw = blockIdx.x * 4 + wv;
  int head = gw & (NHEADS - 1);
  int qt = gw >> 2;
  const float scale = 0.125f;  // 1/sqrt(64), applied to f32 scores

  const _Float16* Qr = Q + (size_t)(qt * 16 + lo) * HDIM + head * HD;
  v8h q0a = *(const v8h*)(Qr + hi * 16);
  v8h q0b = *(const v8h*)(Qr + hi * 16 + 8);
  v8h q1a = *(const v8h*)(Qr + 32 + hi * 16);
  v8h q1b = *(const v8h*)(Qr + 32 + hi * 16 + 8);
  v16h aq0 = __builtin_shufflevector(q0a, q0b, 0,1,2,3,4,5,6,7,8,9,10,11,12,13,14,15);
  v16h aq1 = __builtin_shufflevector(q1a, q1b, 0,1,2,3,4,5,6,7,8,9,10,11,12,13,14,15);

  float mrow[8], lrow[8];
  v8f oacc[4];
#pragma unroll
  for (int r = 0; r < 8; ++r) { mrow[r] = -1e30f; lrow[r] = 0.0f; }
#pragma unroll
  for (int nt = 0; nt < 4; ++nt)
#pragma unroll
    for (int r = 0; r < 8; ++r) oacc[nt][r] = 0.0f;

  for (int k0 = 0; k0 < nk; k0 += 32) {
    // S(16x32) = Qtile(16x64) @ K_chunk^T : 4 WMMAs, each B frag = one 32B vector/lane
    v8f s0 = {0.f,0.f,0.f,0.f,0.f,0.f,0.f,0.f};
    v8f s1 = {0.f,0.f,0.f,0.f,0.f,0.f,0.f,0.f};
    const _Float16* K0 = Kf + (size_t)(k0 + lo) * HDIM + head * HD;
    const _Float16* K1 = Kf + (size_t)(k0 + 16 + lo) * HDIM + head * HD;
    v16h b00 = *(const v16h*)(K0 + hi * 16);
    v16h b01 = *(const v16h*)(K1 + hi * 16);
    v16h b10 = *(const v16h*)(K0 + 32 + hi * 16);
    v16h b11 = *(const v16h*)(K1 + 32 + hi * 16);
    s0 = wmma_f16(aq0, b00, s0);
    s1 = wmma_f16(aq0, b01, s1);
    s0 = wmma_f16(aq1, b10, s0);
    s1 = wmma_f16(aq1, b11, s1);
    // prefetch V fragments now: independent of softmax, loads ride out the VALU work
    v16h bv0 = *(const v16h*)(Vt + (size_t)(head * HD + 0 * 16 + lo) * nk + k0 + hi * 16);
    v16h bv1 = *(const v16h*)(Vt + (size_t)(head * HD + 1 * 16 + lo) * nk + k0 + hi * 16);
    v16h bv2 = *(const v16h*)(Vt + (size_t)(head * HD + 2 * 16 + lo) * nk + k0 + hi * 16);
    v16h bv3 = *(const v16h*)(Vt + (size_t)(head * HD + 3 * 16 + lo) * nk + k0 + hi * 16);
    // online softmax: row stats reduced across the 16 lanes of each half-wave
    float p0[8], p1[8];
#pragma unroll
    for (int r = 0; r < 8; ++r) {
      float a0 = s0[r] * scale, a1 = s1[r] * scale;
      float cm = fmaxf(a0, a1);
#pragma unroll
      for (int m = 1; m < 16; m <<= 1) cm = fmaxf(cm, __shfl_xor(cm, m, 32));
      float nm = fmaxf(mrow[r], cm);
      float corr = __expf(mrow[r] - nm);
      mrow[r] = nm;
      lrow[r] *= corr;
#pragma unroll
      for (int nt = 0; nt < 4; ++nt) oacc[nt][r] *= corr;
      p0[r] = __expf(a0 - nm);
      p1[r] = __expf(a1 - nm);
      float ps = p0[r] + p1[r];
#pragma unroll
      for (int m = 1; m < 16; m <<= 1) ps += __shfl_xor(ps, m, 32);
      lrow[r] += ps;
    }
    // rearrange P from C-layout to A-layout via per-wave LDS tile (16x32 f16 = 1KB)
#pragma unroll
    for (int r = 0; r < 8; ++r) {
      int row = r + 8 * hi;
      plds[wv][row * 32 + lo]      = (_Float16)p0[r];
      plds[wv][row * 32 + 16 + lo] = (_Float16)p1[r];
    }
    __builtin_amdgcn_wave_barrier();   // pin scheduling; same-wave LDS is in-order
    v8h pa = *(const v8h*)&plds[wv][lo * 32 + hi * 16];        // ds_load_b128
    v8h pb = *(const v8h*)&plds[wv][lo * 32 + hi * 16 + 8];
    __builtin_amdgcn_wave_barrier();
    v16h ap = __builtin_shufflevector(pa, pb, 0,1,2,3,4,5,6,7,8,9,10,11,12,13,14,15);
    // O(16x64) += P(16x32) @ V_chunk(32x64): 4 WMMAs with preloaded V frags
    oacc[0] = wmma_f16(ap, bv0, oacc[0]);
    oacc[1] = wmma_f16(ap, bv1, oacc[1]);
    oacc[2] = wmma_f16(ap, bv2, oacc[2]);
    oacc[3] = wmma_f16(ap, bv3, oacc[3]);
  }
  // finalize: divide by softmax denominator, add residual, store
#pragma unroll
  for (int r = 0; r < 8; ++r) {
    float inv = 1.0f / lrow[r];
    int row = qt * 16 + r + 8 * hi;
#pragma unroll
    for (int nt = 0; nt < 4; ++nt) {
      int col = head * HD + nt * 16 + lo;
      Out[(size_t)row * HDIM + col] = Hres[(size_t)row * HDIM + col] + oacc[nt][r] * inv;
    }
  }
}

// ---------------- segment mean (atomic) + head MLP ----------------

__global__ void k_seg_scatter(const float* __restrict__ Hc, const int* __restrict__ batch,
                              float* __restrict__ z, float* __restrict__ cnt,
                              int n, int colbase, int cntbase) {
  int idx = blockIdx.x * blockDim.x + threadIdx.x;
  int node = idx >> 6, q = (idx & 63) << 2;
  if (node >= n) return;
  int g = batch[node];
  float4 hv = *(const float4*)(Hc + (size_t)node * HDIM + q);
  float* zp = z + (size_t)g * (2 * HDIM) + colbase + q;
  atomic_add_f32(zp + 0, hv.x);
  atomic_add_f32(zp + 1, hv.y);
  atomic_add_f32(zp + 2, hv.z);
  atomic_add_f32(zp + 3, hv.w);
  if (q == 0) atomic_add_f32(&cnt[cntbase + g], 1.0f);
}

// per-graph: x = relu(zmean @ fc1_w + fc1_b); out = sigmoid(x @ fc2_w + fc2_b)
__global__ void k_head(const float* __restrict__ z, const float* __restrict__ cnt,
                       const float* __restrict__ w1, const float* __restrict__ b1,
                       const float* __restrict__ w2, const float* __restrict__ b2,
                       float* __restrict__ out) {
  __shared__ float red[256];
  int g = blockIdx.x, t = threadIdx.x;
  float dm = 1.0f / fmaxf(cnt[g], 1.0f);
  float dp = 1.0f / fmaxf(cnt[NG + g], 1.0f);
  float s = b1[t];
  for (int k = 0; k < 2 * HDIM; ++k) {
    float zm = z[(size_t)g * (2 * HDIM) + k] * ((k < HDIM) ? dm : dp);
    s += zm * w1[(size_t)k * HDIM + t];
  }
  s = fmaxf(s, 0.0f);
  red[t] = s * w2[t];
  __syncthreads();
  for (int off = 128; off > 0; off >>= 1) {
    if (t < off) red[t] += red[t + off];
    __syncthreads();
  }
  if (t == 0) out[g] = 1.0f / (1.0f + __expf(-(red[0] + b2[0])));
}

// ---------------- driver ----------------

extern "C" void kernel_launch(void* const* d_in, const int* in_sizes, int n_in,
                              void* d_out, int out_size, void* d_ws, size_t ws_size,
                              hipStream_t stream) {
  (void)in_sizes; (void)n_in; (void)out_size; (void)ws_size;
  const float* x_mol  = (const float*)d_in[0];
  const float* x_prot = (const float*)d_in[1];
  const float* ea_mol = (const float*)d_in[2];
  const float* ea_prot= (const float*)d_in[3];
  const int* ei_mol   = (const int*)d_in[4];
  const int* ei_prot  = (const int*)d_in[5];
  const int* batch_mol  = (const int*)d_in[6];
  const int* batch_prot = (const int*)d_in[7];
  const float* nlm_w = (const float*)d_in[8];  const float* nlm_b = (const float*)d_in[9];
  const float* nlp_w = (const float*)d_in[10]; const float* nlp_b = (const float*)d_in[11];
  const float* elm_w = (const float*)d_in[12]; const float* elm_b = (const float*)d_in[13];
  const float* elp_w = (const float*)d_in[14]; const float* elp_b = (const float*)d_in[15];
  const float* mol_w1 = (const float*)d_in[16]; const float* mol_b1 = (const float*)d_in[17];
  const float* mol_w2 = (const float*)d_in[18]; const float* mol_b2 = (const float*)d_in[19];
  const float* prot_w1 = (const float*)d_in[20]; const float* prot_b1 = (const float*)d_in[21];
  const float* prot_w2 = (const float*)d_in[22]; const float* prot_b2 = (const float*)d_in[23];
  const float* mp_wq = (const float*)d_in[24]; const float* mp_bq = (const float*)d_in[25];
  const float* mp_wk = (const float*)d_in[26]; const float* mp_bk = (const float*)d_in[27];
  const float* mp_wv = (const float*)d_in[28]; const float* mp_bv = (const float*)d_in[29];
  const float* pm_wq = (const float*)d_in[30]; const float* pm_bq = (const float*)d_in[31];
  const float* pm_wk = (const float*)d_in[32]; const float* pm_bk = (const float*)d_in[33];
  const float* pm_wv = (const float*)d_in[34]; const float* pm_bv = (const float*)d_in[35];
  const float* fc1_w = (const float*)d_in[36]; const float* fc1_b = (const float*)d_in[37];
  const float* fc2_w = (const float*)d_in[38]; const float* fc2_b = (const float*)d_in[39];
  float* out = (float*)d_out;

  // ---- workspace carve ----
  float* fws = (float*)d_ws;
  const size_t B = (size_t)NMOL * HDIM;   // 1M elements
  float* hm  = fws + 0 * B;   // mol node features (f32, GINE in-place)
  float* hp  = fws + 1 * B;
  float* agM = fws + 2 * B;   // aggregation buffer; reused as H_mol_c
  float* agP = fws + 3 * B;
  float* tmp = fws + 4 * B;
  _Float16* hws = (_Float16*)(fws + 5 * B);
  _Float16* Qm16  = hws + 0 * B;  // f16 row-major
  _Float16* Kp16  = hws + 1 * B;  // f16 row-major
  _Float16* Vp16t = hws + 2 * B;  // f16 transposed [256, nk]
  _Float16* Qp16  = hws + 3 * B;
  _Float16* Km16  = hws + 4 * B;
  _Float16* Vm16t = hws + 5 * B;
  const size_t WS = (size_t)HDIM * HDIM;  // 65536 per packed weight
  _Float16* wp = hws + 6 * B;             // 14 packed weight slots
  _Float16* p_molw1[2] = { wp + 0 * WS, wp + 2 * WS };
  _Float16* p_molw2[2] = { wp + 1 * WS, wp + 3 * WS };
  _Float16* p_protw1[2] = { wp + 4 * WS, wp + 6 * WS };
  _Float16* p_protw2[2] = { wp + 5 * WS, wp + 7 * WS };
  _Float16* p_mpq = wp + 8 * WS;  _Float16* p_mpk = wp + 9 * WS;  _Float16* p_mpv = wp + 10 * WS;
  _Float16* p_pmq = wp + 11 * WS; _Float16* p_pmk = wp + 12 * WS; _Float16* p_pmv = wp + 13 * WS;
  float* z   = (float*)(wp + 14 * WS);    // [32, 512] pooled sums
  float* cnt = z + NG * 2 * HDIM;         // [2*32]

  // ---- pack all weights into f16 fragment layout (one-shot, L2-resident) ----
  const int PG = (HDIM * HDIM) / 256;
  for (int l = 0; l < 2; ++l) {
    k_pack_w<<<PG, 256, 0, stream>>>(mol_w1 + (size_t)l * WS, p_molw1[l], HDIM, HDIM);
    k_pack_w<<<PG, 256, 0, stream>>>(mol_w2 + (size_t)l * WS, p_molw2[l], HDIM, HDIM);
    k_pack_w<<<PG, 256, 0, stream>>>(prot_w1 + (size_t)l * WS, p_protw1[l], HDIM, HDIM);
    k_pack_w<<<PG, 256, 0, stream>>>(prot_w2 + (size_t)l * WS, p_protw2[l], HDIM, HDIM);
  }
  k_pack_w<<<PG, 256, 0, stream>>>(mp_wq, p_mpq, HDIM, HDIM);
  k_pack_w<<<PG, 256, 0, stream>>>(mp_wk, p_mpk, HDIM, HDIM);
  k_pack_w<<<PG, 256, 0, stream>>>(mp_wv, p_mpv, HDIM, HDIM);
  k_pack_w<<<PG, 256, 0, stream>>>(pm_wq, p_pmq, HDIM, HDIM);
  k_pack_w<<<PG, 256, 0, stream>>>(pm_wk, p_pmk, HDIM, HDIM);
  k_pack_w<<<PG, 256, 0, stream>>>(pm_wv, p_pmv, HDIM, HDIM);

  auto gemm = [&](const float* X, const _Float16* Wp, const float* bias, float* Yf,
                  _Float16* Yh, int nrows, int relu, int omode) {
    int total = (nrows / 16) * (HDIM / 64);
    k_gemm<<<(total + 7) / 8, 256, 0, stream>>>(X, Wp, bias, Yf, Yh, nrows, HDIM, HDIM,
                                                relu, omode);
  };

  // 1) node embeddings
  k_node_embed<<<NMOL * HDIM / 256, 256, 0, stream>>>(x_mol, nlm_w, nlm_b, hm, NMOL, 8);
  k_node_embed<<<NPROT * HDIM / 256, 256, 0, stream>>>(x_prot, nlp_w, nlp_b, hp, NPROT, 4);

  // 2) GINE stacks
  for (int l = 0; l < 2; ++l) {
    k_copy4<<<(int)(B / 4 / 256), 256, 0, stream>>>(hm, agM, (int)(B / 4));
    k_edge_msg<<<EMOL / 4, 256, 0, stream>>>(hm, ea_mol, ei_mol, elm_w, elm_b, agM, EMOL);
    gemm(agM, p_molw1[l], mol_b1 + l * HDIM, tmp, nullptr, NMOL, 1, 0);
    gemm(tmp, p_molw2[l], mol_b2 + l * HDIM, hm, nullptr, NMOL, 1, 0);
  }
  for (int l = 0; l < 2; ++l) {
    k_copy4<<<(int)(B / 4 / 256), 256, 0, stream>>>(hp, agP, (int)(B / 4));
    k_edge_msg<<<EPROT / 4, 256, 0, stream>>>(hp, ea_prot, ei_prot, elp_w, elp_b, agP, EPROT);
    gemm(agP, p_protw1[l], prot_b1 + l * HDIM, tmp, nullptr, NPROT, 1, 0);
    gemm(tmp, p_protw2[l], prot_b2 + l * HDIM, hp, nullptr, NPROT, 1, 0);
  }

  // 3) Q/K/V projections straight to f16 (V transposed for contiguous attention frags)
  gemm(hm, p_mpq, mp_bq, nullptr, Qm16, NMOL, 0, 1);
  gemm(hp, p_mpk, mp_bk, nullptr, Kp16, NPROT, 0, 1);
  gemm(hp, p_mpv, mp_bv, nullptr, Vp16t, NPROT, 0, 2);
  gemm(hp, p_pmq, pm_bq, nullptr, Qp16, NPROT, 0, 1);
  gemm(hm, p_pmk, pm_bk, nullptr, Km16, NMOL, 0, 1);
  gemm(hm, p_pmv, pm_bv, nullptr, Vm16t, NMOL, 0, 2);

  // 4) streaming cross-attention with residual add
  k_attn<<<NMOL / 16, 128, 0, stream>>>(Qm16, Kp16, Vp16t, hm, agM, NMOL, NPROT);
  k_attn<<<NPROT / 16, 128, 0, stream>>>(Qp16, Km16, Vm16t, hp, agP, NPROT, NMOL);

  // 5) segment mean + head
  int zn = NG * 2 * HDIM + 2 * NG;
  k_zero<<<(zn + 255) / 256, 256, 0, stream>>>(z, zn);
  k_seg_scatter<<<NMOL / 4, 256, 0, stream>>>(agM, batch_mol, z, cnt, NMOL, 0, 0);
  k_seg_scatter<<<NPROT / 4, 256, 0, stream>>>(agP, batch_prot, z, cnt, NPROT, HDIM, NG);
  k_head<<<NG, 256, 0, stream>>>(z, cnt, fc1_w, fc1_b, fc2_w, fc2_b, out);
}
